// JK_GAMLP_62723702391587
// MI455X (gfx1250) — matrix operand order repaired and checked by reading, not compiled, not run on hardware
//
#include <hip/hip_runtime.h>

typedef _Float16 v16h __attribute__((ext_vector_type(16)));
typedef _Float16 v8h  __attribute__((ext_vector_type(8)));
typedef float    v8f  __attribute__((ext_vector_type(8)));
typedef float    v4f  __attribute__((ext_vector_type(4)));

// Problem constants
static constexpr int kL = 8;
static constexpr int kN = 100000;
static constexpr int kH = 128;
static constexpr int kK = kL * kH;      // 1024
static constexpr int kNT = 32;          // n-rows per block (two 16-row M-tiles)
static constexpr int kKSTEPS = kK / 32; // 32 WMMA steps of K=32
static constexpr int kFragHalves = 32 * 32 * 16; // one M-tile of A fragments

// ---------------------------------------------------------------------------
// Kernel 0: pack w_ref (H=128 rows x K=1024 cols, row-major f32) into f16
// B-matrix fragments for v_wmma_f32_16x16x32_f16.
// B fragment for (colblock c, kstep t): 32 lanes x 16 halves.
//   lane n (<16)  : halves h = B[K=32t+h   ][N=16c+n] = w_ref[(16c+n)*1024 + 32t+h]
//   lane n+16     : halves h = B[K=32t+16+h][N=16c+n]
// Flat layout: bpack[(((c*32)+t)*32 + lane)*16 + half]
// ---------------------------------------------------------------------------
__global__ void pack_w_kernel(const float* __restrict__ w_ref,
                              _Float16* __restrict__ bpack) {
    int idx  = blockIdx.x * blockDim.x + threadIdx.x;   // 8*32*32 = 8192 threads
    int lane = idx & 31;
    int t    = (idx >> 5) & 31;
    int c    = idx >> 10;
    int col  = c * 16 + (lane & 15);
    int kb   = t * 32 + ((lane & 16) ? 16 : 0);
    const float* src = w_ref + (size_t)col * kK + kb;
    v8h lo, hi;
#pragma unroll
    for (int j = 0; j < 8; ++j) lo[j] = (_Float16)src[j];
#pragma unroll
    for (int j = 0; j < 8; ++j) hi[j] = (_Float16)src[8 + j];
    _Float16* dst = bpack + (size_t)idx * 16;
    *(v8h*)dst       = lo;
    *(v8h*)(dst + 8) = hi;
}

// ---------------------------------------------------------------------------
// Kernel 1: fused GAMLP block. 256 threads = 8 waves; block handles 32 n-rows
// as two 16-row M-tiles sharing each B fragment (2 WMMA per B load).
// ---------------------------------------------------------------------------
__global__ void __launch_bounds__(256, 1)
gamlp_fused_kernel(const float* __restrict__ xs,
                   const _Float16* __restrict__ bpack,
                   const float* __restrict__ b_ref,
                   const float* __restrict__ w_att,
                   const float* __restrict__ b_att_p,
                   const float* __restrict__ alpha_p,
                   float* __restrict__ out) {
    // A fragments in LDS: 2 M-tiles x (32 ksteps x 32 lanes x 16 halves) = 64 KB
    __shared__ alignas(32) _Float16 Alds[2 * kFragHalves];
    __shared__ float sX[kL][kNT];   // xs . wa_x  per (l, m)
    __shared__ float sJK[kNT];      // jk . wa_ref per m

    const int tid = threadIdx.x;
    const int n0  = blockIdx.x * kNT;

    if (tid < kNT) sJK[tid] = 0.0f;

    // ---- Phase 1: load xs tile, convert->f16 in A-fragment layout, dot wa_x
    const int hg = tid & 15;        // h-group: covers h = hg*8 .. hg*8+7
    const int h8 = hg * 8;
    const int g  = tid >> 4;        // 16 threads per (l,m) pair

    float wax[8];
#pragma unroll
    for (int j = 0; j < 8; ++j) wax[j] = w_att[kH + h8 + j];

#pragma unroll
    for (int rep = 0; rep < 16; ++rep) {
        const int lm = g + rep * 16;            // 0..255
        const int l  = lm >> 5;
        const int m  = lm & 31;
        const float* src = xs + ((size_t)l * kN + (n0 + m)) * kH + h8;
        v4f a = *(const v4f*)src;
        v4f b = *(const v4f*)(src + 4);
        float dot = 0.0f;
        v8h hv;
#pragma unroll
        for (int j = 0; j < 4; ++j) { dot += a[j] * wax[j];     hv[j]     = (_Float16)a[j]; }
#pragma unroll
        for (int j = 0; j < 4; ++j) { dot += b[j] * wax[4 + j]; hv[4 + j] = (_Float16)b[j]; }
        // reduce over the 16 lanes sharing this (l,m)  (xor<=8 stays in half-wave)
#pragma unroll
        for (int d = 1; d < 16; d <<= 1) dot += __shfl_xor(dot, d, 32);
        if (hg == 0) sX[l][m] = dot;
        // scatter 8 halves into A-fragment layout
        const int mt   = m >> 4;                 // which M-tile
        const int mr   = m & 15;                 // row within tile
        const int k    = l * kH + h8;
        const int t    = k >> 5;
        const int r0   = k & 31;                 // 0,8,16,24
        const int lane = mr + ((r0 & 16) ? 16 : 0);
        const int hb   = (r0 & 8) ? 8 : 0;
        *(v8h*)&Alds[mt * kFragHalves + (t * 32 + lane) * 16 + hb] = hv;
    }
    __syncthreads();

    // ---- Phase 2: GEMM via WMMA. Wave w owns output columns [16w, 16w+16).
    const int wv   = tid >> 5;
    const int lane = tid & 31;
    v8f c0, c1;
#pragma unroll
    for (int r = 0; r < 8; ++r) { c0[r] = 0.0f; c1[r] = 0.0f; }

    const _Float16* bp = bpack + ((size_t)(wv * 32) * 32 + lane) * 16;
#pragma unroll 4
    for (int t = 0; t < kKSTEPS; ++t) {
        v16h bv  = *(const v16h*)(bp + (size_t)t * 32 * 16);
        v16h av0 = *(const v16h*)&Alds[(t * 32 + lane) * 16];
        v16h av1 = *(const v16h*)&Alds[kFragHalves + (t * 32 + lane) * 16];
        c0 = __builtin_amdgcn_wmma_f32_16x16x32_f16(
                 false, av0, false, bv, (short)0, c0, false, false);
        c1 = __builtin_amdgcn_wmma_f32_16x16x32_f16(
                 false, av1, false, bv, (short)0, c1, false, false);
    }

    // epilogue: + b_ref, PReLU, dot with wa_ref, reduce into sJK[m]
    {
        const int   col   = wv * 16 + (lane & 15);
        const float br    = b_ref[col];
        const float war   = w_att[col];
        const float alpha = *alpha_p;
#pragma unroll
        for (int r = 0; r < 8; ++r) {
            float v0 = c0[r] + br;
            float v1 = c1[r] + br;
            v0 = (v0 >= 0.0f) ? v0 : alpha * v0;
            v1 = (v1 >= 0.0f) ? v1 : alpha * v1;
            float p0 = v0 * war;
            float p1 = v1 * war;
#pragma unroll
            for (int d = 1; d < 16; d <<= 1) {
                p0 += __shfl_xor(p0, d, 32);
                p1 += __shfl_xor(p1, d, 32);
            }
            if ((lane & 15) == 0) {
                const int m = r + ((lane & 16) ? 8 : 0);   // C/D: lanes>=16 -> M=r+8
                atomicAdd(&sJK[m],      p0);
                atomicAdd(&sJK[m + 16], p1);
            }
        }
    }
    __syncthreads();

    // ---- Phase 3: scores -> softmax over l -> weighted sum of xs
    const float batt = *b_att_p;
#pragma unroll
    for (int sub = 0; sub < 2; ++sub) {
        const int m = (tid >> 4) + sub * 16;    // row within tile (0..31)
        const float sjk = sJK[m];
        float sc[kL];
        float mx = -1e30f;
#pragma unroll
        for (int l = 0; l < kL; ++l) {
            float s = sjk + sX[l][m] + batt;
            s = (s > 0.0f) ? s : 0.0f;          // relu
            sc[l] = s;
            mx = fmaxf(mx, s);
        }
        float sum = 0.0f;
#pragma unroll
        for (int l = 0; l < kL; ++l) { sc[l] = __expf(sc[l] - mx); sum += sc[l]; }
        const float inv = 1.0f / sum;

        float acc[8];
#pragma unroll
        for (int j = 0; j < 8; ++j) acc[j] = 0.0f;
#pragma unroll
        for (int l = 0; l < kL; ++l) {
            const float* src = xs + ((size_t)l * kN + (n0 + m)) * kH + h8;
            v4f a = *(const v4f*)src;
            v4f b = *(const v4f*)(src + 4);
            const float w = sc[l] * inv;
#pragma unroll
            for (int j = 0; j < 4; ++j) { acc[j] += w * a[j]; acc[4 + j] += w * b[j]; }
        }
        float* dst = out + (size_t)(n0 + m) * kH + h8;
        v4f o0, o1;
#pragma unroll
        for (int j = 0; j < 4; ++j) { o0[j] = acc[j]; o1[j] = acc[4 + j]; }
        *(v4f*)dst       = o0;
        *(v4f*)(dst + 4) = o1;
    }
}

// ---------------------------------------------------------------------------
extern "C" void kernel_launch(void* const* d_in, const int* in_sizes, int n_in,
                              void* d_out, int out_size, void* d_ws, size_t ws_size,
                              hipStream_t stream) {
    const float* xs      = (const float*)d_in[0];   // (L, N, H)
    const float* w_ref   = (const float*)d_in[1];   // (H, L*H)
    const float* b_ref   = (const float*)d_in[2];   // (H,)
    const float* w_att   = (const float*)d_in[3];   // (2H,)
    const float* b_att   = (const float*)d_in[4];   // scalar
    const float* alpha   = (const float*)d_in[5];   // scalar
    float*       out     = (float*)d_out;           // (N, H)

    _Float16* bpack = (_Float16*)d_ws;              // 1024*128 f16 = 256 KB

    // pack w_ref into WMMA B fragments (8*32*32 = 8192 threads)
    pack_w_kernel<<<32, 256, 0, stream>>>(w_ref, bpack);

    // fused GAMLP: 100000 / 32 = 3125 blocks
    gamlp_fused_kernel<<<kN / kNT, 256, 0, stream>>>(
        xs, bpack, b_ref, w_att, b_att, alpha, out);
}
